// RetinaFace_23922967839081
// MI455X (gfx1250) — matrix operand order repaired and checked by reading, not compiled, not run on hardware
//
#include <hip/hip_runtime.h>
#include <hip/hip_bf16.h>

typedef __bf16 bf16;
typedef __attribute__((ext_vector_type(16))) __bf16 v16bf;
typedef __attribute__((ext_vector_type(8)))  float  v8f;
typedef __attribute__((ext_vector_type(4)))  unsigned int u32x4;

#define NIMG 8
#define NPRI 16800
#define TOPK 2048
#define SORTN 32768
#define SCORE_THR 0.5f
#define IOU_THR 0.4f

// ---------------- layout conversion: NCHW fp32 -> NHWC bf16 ----------------
__global__ void nchw2nhwc_kernel(const float* __restrict__ in, bf16* __restrict__ out,
                                 int C, int HW) {
  size_t idx = (size_t)blockIdx.x * blockDim.x + threadIdx.x;
  size_t total = (size_t)NIMG * C * HW;
  if (idx >= total) return;
  int p = (int)(idx % HW);
  size_t r = idx / HW;
  int c = (int)(r % C);
  int n = (int)(r / C);
  out[((size_t)n * HW + p) * C + c] = (bf16)in[idx];
}

// ---------------- weight packing: [OC][IC][T] fp32 -> [T][OC][IC] bf16 -----
__global__ void pack_w_kernel(const float* __restrict__ w, bf16* __restrict__ wp,
                              int OC, int IC, int T) {
  size_t idx = (size_t)blockIdx.x * blockDim.x + threadIdx.x;
  size_t total = (size_t)OC * IC * T;
  if (idx >= total) return;
  int t = (int)(idx % T);
  size_t r = idx / T;
  int ic = (int)(r % IC);
  int oc = (int)(r / IC);
  wp[((size_t)t * OC + oc) * IC + ic] = (bf16)w[idx];
}

// ---------------- generic implicit-GEMM conv via BF16 WMMA -----------------
// One wave computes a 32(OC) x 32(pixel) macro-tile: 2x2 register blocking,
// 4 WMMAs per K-step reusing 2 A- and 2 B-fragments. 3x3 convs run as 9
// shifted 1x1 GEMMs accumulating into the same F32 fragments.
__global__ __launch_bounds__(32) void conv_wmma_kernel(
    const bf16* __restrict__ X,   // [NIMG][H*W][IC] NHWC
    const bf16* __restrict__ Wp,  // [taps][OC][IC]
    const float* __restrict__ bias,
    bf16* __restrict__ Y,         // [NIMG][H*W][OCtot], write at chanOff
    int H, int W, int IC, int OC, int OCtot, int chanOff, int taps, int relu)
{
  const int lane = threadIdx.x;
  const int hi = lane >> 4;      // 0: lanes 0-15, 1: lanes 16-31
  const int lo = lane & 15;
  const int HW = H * W;
  const int n0 = blockIdx.x * 32;          // pixel macro-tile base
  const int m0 = blockIdx.y * 32;          // OC macro-tile base (OC % 32 == 0)
  const int img = blockIdx.z;
  const bool hasN1 = (n0 + 16) < HW;       // wave-uniform: second pixel tile present

  const bf16* Xi = X + (size_t)img * HW * IC;
  const int p0 = n0 + lo;                  // column pixel, tile 0
  const int p1 = n0 + 16 + lo;             // column pixel, tile 1
  const int py0 = p0 / W, px0 = p0 % W;
  const int py1 = hasN1 ? (p1 / W) : 0;
  const int px1 = hasN1 ? (p1 % W) : 0;

  v8f c00, c01, c10, c11;
#pragma unroll
  for (int j = 0; j < 8; ++j) { c00[j] = 0.0f; c01[j] = 0.0f; c10[j] = 0.0f; c11[j] = 0.0f; }

  const bf16* arow0 = Wp + (size_t)(m0 + lo) * IC;        // OC rows m0..m0+15
  const bf16* arow1 = Wp + (size_t)(m0 + 16 + lo) * IC;   // OC rows m0+16..m0+31

  for (int t = 0; t < taps; ++t) {
    int dy = 0, dx = 0;
    if (taps == 9) { dy = t / 3 - 1; dx = t % 3 - 1; }
    const int sy0 = py0 + dy, sx0 = px0 + dx;
    const int sy1 = py1 + dy, sx1 = px1 + dx;
    const bool v0 = (sy0 >= 0) && (sy0 < H) && (sx0 >= 0) && (sx0 < W);
    const bool v1 = hasN1 && (sy1 >= 0) && (sy1 < H) && (sx1 >= 0) && (sx1 < W);
    const bf16* brow0 = Xi + (size_t)(sy0 * W + sx0) * IC;
    const bf16* brow1 = Xi + (size_t)(sy1 * W + sx1) * IC;
    const size_t toff = (size_t)t * OC * IC;
    for (int kb = 0; kb < IC; kb += 32) {
      union { v16bf v; u32x4 q[2]; } a0, a1, b0, b1;
      // A (16x32 bf16): lanes 0-15 -> K {kb..kb+7, kb+16..kb+23}
      //                 lanes16-31 -> K {kb+8..15, kb+24..31}
      a0.q[0] = *(const u32x4*)(arow0 + toff + kb + hi * 8);
      a0.q[1] = *(const u32x4*)(arow0 + toff + kb + 16 + hi * 8);
      a1.q[0] = *(const u32x4*)(arow1 + toff + kb + hi * 8);
      a1.q[1] = *(const u32x4*)(arow1 + toff + kb + 16 + hi * 8);
      // B (32x16 bf16): lanes 0-15 -> K {kb..kb+7, kb+16..kb+23} ; lanes16-31 -> K {kb+8..15, kb+24..31}
      u32x4 z = {0u, 0u, 0u, 0u};
      if (v0) {
        b0.q[0] = *(const u32x4*)(brow0 + kb + hi * 8);
        b0.q[1] = *(const u32x4*)(brow0 + kb + 16 + hi * 8);
      } else { b0.q[0] = z; b0.q[1] = z; }
      c00 = __builtin_amdgcn_wmma_f32_16x16x32_bf16(false, a0.v, false, b0.v,
                                                    (short)0, c00, false, false);
      c10 = __builtin_amdgcn_wmma_f32_16x16x32_bf16(false, a1.v, false, b0.v,
                                                    (short)0, c10, false, false);
      if (hasN1) {  // uniform branch: EXEC stays all-1s for the WMMAs inside
        if (v1) {
          b1.q[0] = *(const u32x4*)(brow1 + kb + hi * 8);
          b1.q[1] = *(const u32x4*)(brow1 + kb + 16 + hi * 8);
        } else { b1.q[0] = z; b1.q[1] = z; }
        c01 = __builtin_amdgcn_wmma_f32_16x16x32_bf16(false, a0.v, false, b1.v,
                                                      (short)0, c01, false, false);
        c11 = __builtin_amdgcn_wmma_f32_16x16x32_bf16(false, a1.v, false, b1.v,
                                                      (short)0, c11, false, false);
      }
    }
  }

  // C/D layout: VGPR j holds OC row mBase + hi*8 + j, pixel column nBase+lo.
  const float* bias0 = bias + m0 + hi * 8;
  const float* bias1 = bias + m0 + 16 + hi * 8;
  union { bf16 h[8]; u32x4 q; } o;
  bf16* y0 = Y + ((size_t)img * HW + p0) * OCtot + chanOff + m0 + hi * 8;
#pragma unroll
  for (int j = 0; j < 8; ++j) {
    float v = c00[j] + bias0[j];
    if (relu) v = fmaxf(v, 0.0f);
    o.h[j] = (bf16)v;
  }
  *(u32x4*)y0 = o.q;
#pragma unroll
  for (int j = 0; j < 8; ++j) {
    float v = c10[j] + bias1[j];
    if (relu) v = fmaxf(v, 0.0f);
    o.h[j] = (bf16)v;
  }
  *(u32x4*)(y0 + 16) = o.q;
  if (hasN1) {
    bf16* y1 = Y + ((size_t)img * HW + p1) * OCtot + chanOff + m0 + hi * 8;
#pragma unroll
    for (int j = 0; j < 8; ++j) {
      float v = c01[j] + bias0[j];
      if (relu) v = fmaxf(v, 0.0f);
      o.h[j] = (bf16)v;
    }
    *(u32x4*)y1 = o.q;
#pragma unroll
    for (int j = 0; j < 8; ++j) {
      float v = c11[j] + bias1[j];
      if (relu) v = fmaxf(v, 0.0f);
      o.h[j] = (bf16)v;
    }
    *(u32x4*)(y1 + 16) = o.q;
  }
}

// ---------------- FPN lateral add with nearest 2x upsample -----------------
__global__ void add_up2_kernel(const bf16* __restrict__ A, const bf16* __restrict__ Bs,
                               bf16* __restrict__ O, int H, int W) {
  size_t idx = (size_t)blockIdx.x * blockDim.x + threadIdx.x;
  size_t total = (size_t)NIMG * H * W * 256;
  if (idx >= total) return;
  int c = (int)(idx & 255);
  size_t pi = idx >> 8;
  int p = (int)(pi % (size_t)(H * W));
  int img = (int)(pi / (size_t)(H * W));
  int y = p / W, x = p % W;
  int hw2 = (H / 2) * (W / 2);
  float v = (float)A[idx] +
            (float)Bs[(((size_t)img * hw2 + (size_t)(y / 2) * (W / 2) + (x / 2)) << 8) + c];
  O[idx] = (bf16)v;
}

// ---------------- heads + softmax + prior decode + key pack ----------------
__global__ void head_decode_kernel(
    const bf16* __restrict__ F, const float* __restrict__ Wb, const float* __restrict__ Bb,
    const float* __restrict__ Wc, const float* __restrict__ Bc,
    float* __restrict__ boxes, unsigned long long* __restrict__ keys,
    int H, int W, int stridePx, float anc0, float anc1, int gbase)
{
  int idx = blockIdx.x * blockDim.x + threadIdx.x;
  int HW = H * W;
  if (idx >= NIMG * HW) return;
  int img = idx / HW, p = idx % HW;
  const bf16* f = F + ((size_t)img * HW + p) * 256;
  float acc[12];
#pragma unroll
  for (int j = 0; j < 8; ++j) acc[j] = Bb[j];
#pragma unroll
  for (int j = 0; j < 4; ++j) acc[8 + j] = Bc[j];
  for (int ic = 0; ic < 256; ++ic) {
    float xv = (float)f[ic];
#pragma unroll
    for (int j = 0; j < 8; ++j) acc[j] += xv * Wb[j * 256 + ic];
#pragma unroll
    for (int j = 0; j < 4; ++j) acc[8 + j] += xv * Wc[j * 256 + ic];
  }
  float pcx = (float)((p % W) * stridePx + stridePx / 2);
  float pcy = (float)((p / W) * stridePx + stridePx / 2);
  float anc[2] = {anc0, anc1};
#pragma unroll
  for (int a = 0; a < 2; ++a) {
    float pw = anc[a];
    float cx = pw * 0.1f * acc[a * 4 + 0] + pcx;
    float cy = pw * 0.1f * acc[a * 4 + 1] + pcy;
    float w2 = pw * expf(0.2f * acc[a * 4 + 2]) * 0.5f;
    float h2 = pw * expf(0.2f * acc[a * 4 + 3]) * 0.5f;
    float l0 = acc[8 + a * 2], l1 = acc[8 + a * 2 + 1];
    float s = 1.0f / (1.0f + expf(l0 - l1));  // softmax[...,1]
    int gi = gbase + p * 2 + a;
    float* bo = boxes + ((size_t)img * NPRI + gi) * 4;
    bo[0] = cx - w2; bo[1] = cy - h2; bo[2] = cx + w2; bo[3] = cy + h2;
    float st = (s > SCORE_THR) ? s : -1.0f;
    unsigned int u = __float_as_uint(st);
    u = (u & 0x80000000u) ? ~u : (u | 0x80000000u);  // monotone float->uint
    keys[(size_t)img * SORTN + gi] =
        ((unsigned long long)u << 32) | (unsigned long long)(0xFFFFFFFFu - (unsigned)gi);
  }
}

__global__ void fill_u64_kernel(unsigned long long* __restrict__ p, size_t n) {
  size_t i = (size_t)blockIdx.x * blockDim.x + threadIdx.x;
  if (i < n) p[i] = 0ull;
}

// ---------------- bitonic sort step (descending by 64-bit key) -------------
__global__ void bitonic_step_kernel(unsigned long long* __restrict__ keys, int j, int k) {
  int gid = blockIdx.x * blockDim.x + threadIdx.x;
  if (gid >= NIMG * SORTN) return;
  int img = gid >> 15;
  int t = gid & (SORTN - 1);
  int ixj = t ^ j;
  if (ixj <= t) return;
  unsigned long long* a = keys + (size_t)img * SORTN;
  unsigned long long x = a[t], y = a[ixj];
  bool up = (t & k) != 0;  // ascending subsequence -> overall descending
  bool sw = up ? (x > y) : (x < y);
  if (sw) { a[t] = y; a[ixj] = x; }
}

// ---------------- greedy NMS, one block per image --------------------------
__global__ __launch_bounds__(256) void nms_kernel(
    const unsigned long long* __restrict__ keys, const float* __restrict__ boxes,
    float* __restrict__ dets, float* __restrict__ keepOut)
{
  __shared__ float bx[TOPK], by[TOPK], bX[TOPK], bY[TOPK], sc[TOPK];
  __shared__ unsigned char keep[TOPK];
  int img = blockIdx.x;
  const unsigned long long* kk = keys + (size_t)img * SORTN;
  const float* bb = boxes + (size_t)img * NPRI * 4;
  for (int i = threadIdx.x; i < TOPK; i += blockDim.x) {
    unsigned long long key = kk[i];
    unsigned int u = (unsigned int)(key >> 32);
    float s = (u & 0x80000000u) ? __uint_as_float(u & 0x7FFFFFFFu) : __uint_as_float(~u);
    unsigned int gi = 0xFFFFFFFFu - (unsigned int)(key & 0xFFFFFFFFu);
    if (gi >= NPRI) gi = 0;
    bx[i] = bb[gi * 4 + 0]; by[i] = bb[gi * 4 + 1];
    bX[i] = bb[gi * 4 + 2]; bY[i] = bb[gi * 4 + 3];
    sc[i] = s;
    keep[i] = (s > SCORE_THR) ? 1 : 0;
  }
  __syncthreads();
  for (int i = 0; i < TOPK; ++i) {
    if (keep[i]) {  // uniform branch (shared read)
      float x1 = bx[i], y1 = by[i], x2 = bX[i], y2 = bY[i];
      float ai = (x2 - x1) * (y2 - y1);
      for (int j = i + 1 + threadIdx.x; j < TOPK; j += blockDim.x) {
        if (!keep[j]) continue;
        float lx = fmaxf(x1, bx[j]), ly = fmaxf(y1, by[j]);
        float rx = fminf(x2, bX[j]), ry = fminf(y2, bY[j]);
        float w = fmaxf(rx - lx, 0.0f), h = fmaxf(ry - ly, 0.0f);
        float inter = w * h;
        float aj = (bX[j] - bx[j]) * (bY[j] - by[j]);
        float iou = inter / (ai + aj - inter + 1e-9f);
        if (iou > IOU_THR) keep[j] = 0;
      }
    }
    __syncthreads();
  }
  for (int i = threadIdx.x; i < TOPK; i += blockDim.x) {
    float m = keep[i] ? 1.0f : 0.0f;
    float* d = dets + ((size_t)img * TOPK + i) * 5;
    d[0] = bx[i] * m; d[1] = by[i] * m; d[2] = bX[i] * m; d[3] = bY[i] * m; d[4] = sc[i] * m;
    keepOut[(size_t)img * TOPK + i] = m;
  }
}

// =========================== host orchestration ============================
extern "C" void kernel_launch(void* const* d_in, const int* in_sizes, int n_in,
                              void* d_out, int out_size, void* d_ws, size_t ws_size,
                              hipStream_t stream)
{
  if (n_in < 55) return;
  const float *x[3];
  const float *foW[3], *foB[3], *mgW[2], *mgB[2];
  const float *sshW[3][5], *sshB[3][5];
  const float *hbW[3], *hbB[3], *hcW[3], *hcB[3];
  auto Fp = [&](int i) { return (const float*)d_in[i]; };

  if (in_sizes[0] == 8 * 512 * 80 * 80) {
    // insertion-order flattening: x1,x2,x3 then params (w before b)
    x[0] = Fp(0); x[1] = Fp(1); x[2] = Fp(2);
    int i = 3;
    for (int k = 0; k < 3; ++k) { foW[k] = Fp(i++); foB[k] = Fp(i++); }
    for (int k = 0; k < 2; ++k) { mgW[k] = Fp(i++); mgB[k] = Fp(i++); }
    for (int m = 0; m < 3; ++m)
      for (int c = 0; c < 5; ++c) { sshW[m][c] = Fp(i++); sshB[m][c] = Fp(i++); }
    for (int k = 0; k < 3; ++k) { hbW[k] = Fp(i++); hbB[k] = Fp(i++); }
    for (int k = 0; k < 3; ++k) { hcW[k] = Fp(i++); hcB[k] = Fp(i++); }
  } else {
    // jax pytree flattening: dict keys sorted alphabetically, b before w
    int i = 0;
    for (int k = 0; k < 2; ++k) { mgB[k] = Fp(i++); mgW[k] = Fp(i++); }
    for (int k = 0; k < 3; ++k) { foB[k] = Fp(i++); foW[k] = Fp(i++); }
    for (int k = 0; k < 3; ++k) { hbB[k] = Fp(i++); hbW[k] = Fp(i++); }
    for (int k = 0; k < 3; ++k) { hcB[k] = Fp(i++); hcW[k] = Fp(i++); }
    for (int m = 0; m < 3; ++m)
      for (int c = 0; c < 5; ++c) { sshB[m][c] = Fp(i++); sshW[m][c] = Fp(i++); }
    x[0] = Fp(i++); x[1] = Fp(i++); x[2] = Fp(i++);
  }

  // ---------------- workspace layout (bytes) ----------------
  const size_t OFF_XB1 = 0;                       // 8*6400*512*2
  const size_t OFF_XB2 = 52428800;                // 8*1600*1024*2
  const size_t OFF_XB3 = 78643200;                // 8*400*2048*2
  const size_t OFF_YS0 = 91750400;                // 8*6400*256*2
  const size_t OFF_YS1 = 117964800;               // 8*1600*256*2
  const size_t OFF_YS2 = 124518400;               // 8*400*256*2
  const size_t OFF_FS1 = 126156800;
  const size_t OFF_FS2 = 132710400;
  const size_t OFF_T64 = 134348800;               // 8*6400*64*2 (shared scratch)
  const size_t OFF_R64 = 140902400;
  const size_t OFF_WPK = 147456000;               // packed bf16 weights
  const size_t OFF_BOX = 154968064;               // 8*16800*4*4
  const size_t OFF_KEY = 157118464;               // 8*32768*8
  const size_t WS_NEED = 159215616;
  if (ws_size < WS_NEED) return;
  // reuse: after fpn_out convs the NHWC input copies are dead
  const size_t OFF_YSUM0 = OFF_XB1;               // 26,214,400
  const size_t OFF_YS0M  = OFF_XB1 + 26214400;    // 26,214,400
  const size_t OFF_FS0   = OFF_XB2;               // 26,214,400
  const size_t OFF_YSUM1 = OFF_XB3;               //  6,553,600
  const size_t OFF_YS1M  = OFF_XB3 + 6553600;     //  6,553,600

  char* ws = (char*)d_ws;
  auto B = [&](size_t off) { return (bf16*)(ws + off); };

  // packed weight offsets (bytes, relative to OFF_WPK)
  const size_t WFO0 = OFF_WPK + 0;                // 256*512*2
  const size_t WFO1 = OFF_WPK + 262144;           // 256*1024*2
  const size_t WFO2 = OFF_WPK + 786432;           // 256*2048*2
  const size_t WM0  = OFF_WPK + 1835008;          // 9*256*256*2
  const size_t WM1  = OFF_WPK + 3014656;
  const size_t WSSH = OFF_WPK + 4194304;          // 3 modules x 1,105,920
  const size_t SSH_MOD = 1105920;
  const size_t SSH_C[5] = {0, 589824, 884736, 958464, 1032192};

  // ---------------- 1) NHWC bf16 input copies ----------------
  {
    size_t n = (size_t)NIMG * 512 * 6400;
    nchw2nhwc_kernel<<<dim3((unsigned)((n + 255) / 256)), 256, 0, stream>>>(x[0], B(OFF_XB1), 512, 6400);
    n = (size_t)NIMG * 1024 * 1600;
    nchw2nhwc_kernel<<<dim3((unsigned)((n + 255) / 256)), 256, 0, stream>>>(x[1], B(OFF_XB2), 1024, 1600);
    n = (size_t)NIMG * 2048 * 400;
    nchw2nhwc_kernel<<<dim3((unsigned)((n + 255) / 256)), 256, 0, stream>>>(x[2], B(OFF_XB3), 2048, 400);
  }

  // ---------------- 2) pack weights ----------------
  auto pack = [&](const float* w, size_t off, int OC, int IC, int T) {
    size_t n = (size_t)OC * IC * T;
    pack_w_kernel<<<dim3((unsigned)((n + 255) / 256)), 256, 0, stream>>>(w, B(off), OC, IC, T);
  };
  pack(foW[0], WFO0, 256, 512, 1);
  pack(foW[1], WFO1, 256, 1024, 1);
  pack(foW[2], WFO2, 256, 2048, 1);
  pack(mgW[0], WM0, 256, 256, 9);
  pack(mgW[1], WM1, 256, 256, 9);
  for (int m = 0; m < 3; ++m) {
    size_t base = WSSH + (size_t)m * SSH_MOD;
    pack(sshW[m][0], base + SSH_C[0], 128, 256, 9);
    pack(sshW[m][1], base + SSH_C[1], 64, 256, 9);
    pack(sshW[m][2], base + SSH_C[2], 64, 64, 9);
    pack(sshW[m][3], base + SSH_C[3], 64, 64, 9);
    pack(sshW[m][4], base + SSH_C[4], 64, 64, 9);
  }

  // ---------------- 3) convs (WMMA, 32x32 macro-tiles) ----------------
  auto conv = [&](const bf16* X, size_t woff, const float* bias, bf16* Y,
                  int H, int W, int IC, int OC, int OCtot, int chOff, int taps, int relu) {
    dim3 g((unsigned)((H * W + 31) / 32), (unsigned)(OC / 32), NIMG);
    conv_wmma_kernel<<<g, 32, 0, stream>>>(X, B(woff), bias, Y, H, W, IC, OC, OCtot, chOff, taps, relu);
  };

  // FPN lateral 1x1 convs (relu)
  conv(B(OFF_XB3), WFO2, foB[2], B(OFF_YS2), 20, 20, 2048, 256, 256, 0, 1, 1);
  conv(B(OFF_XB2), WFO1, foB[1], B(OFF_YS1), 40, 40, 1024, 256, 256, 0, 1, 1);
  conv(B(OFF_XB1), WFO0, foB[0], B(OFF_YS0), 80, 80,  512, 256, 256, 0, 1, 1);

  // top-down merges
  {
    size_t n = (size_t)NIMG * 1600 * 256;
    add_up2_kernel<<<dim3((unsigned)((n + 255) / 256)), 256, 0, stream>>>(B(OFF_YS1), B(OFF_YS2), B(OFF_YSUM1), 40, 40);
  }
  conv(B(OFF_YSUM1), WM1, mgB[1], B(OFF_YS1M), 40, 40, 256, 256, 256, 0, 9, 1);
  {
    size_t n = (size_t)NIMG * 6400 * 256;
    add_up2_kernel<<<dim3((unsigned)((n + 255) / 256)), 256, 0, stream>>>(B(OFF_YS0), B(OFF_YS1M), B(OFF_YSUM0), 80, 80);
  }
  conv(B(OFF_YSUM0), WM0, mgB[0], B(OFF_YS0M), 80, 80, 256, 256, 256, 0, 9, 1);

  // SSH modules (final relu fused per concat slice)
  const size_t inOff[3]  = {OFF_YS0M, OFF_YS1M, OFF_YS2};
  const size_t outOff[3] = {OFF_FS0, OFF_FS1, OFF_FS2};
  const int    lh[3] = {80, 40, 20};
  for (int l = 0; l < 3; ++l) {
    size_t wb = WSSH + (size_t)l * SSH_MOD;
    int H = lh[l], W = lh[l];
    conv(B(inOff[l]), wb + SSH_C[0], sshB[l][0], B(outOff[l]), H, W, 256, 128, 256,   0, 9, 1); // c1 -> ch 0..127
    conv(B(inOff[l]), wb + SSH_C[1], sshB[l][1], B(OFF_T64),  H, W, 256,  64,  64,   0, 9, 1); // t = relu(c2)
    conv(B(OFF_T64),  wb + SSH_C[2], sshB[l][2], B(outOff[l]), H, W,  64,  64, 256, 128, 9, 1); // c3 -> ch 128..191
    conv(B(OFF_T64),  wb + SSH_C[3], sshB[l][3], B(OFF_R64),  H, W,  64,  64,  64,   0, 9, 1); // r = relu(c4)
    conv(B(OFF_R64),  wb + SSH_C[4], sshB[l][4], B(outOff[l]), H, W,  64,  64, 256, 192, 9, 1); // c5 -> ch 192..255
  }

  // ---------------- 4) heads + decode + key pack ----------------
  float* boxes = (float*)(ws + OFF_BOX);
  unsigned long long* keys = (unsigned long long*)(ws + OFF_KEY);
  {
    size_t n = (size_t)NIMG * SORTN;
    fill_u64_kernel<<<dim3((unsigned)((n + 255) / 256)), 256, 0, stream>>>(keys, n);
  }
  head_decode_kernel<<<dim3((NIMG * 6400 + 255) / 256), 256, 0, stream>>>(
      B(OFF_FS0), hbW[0], hbB[0], hcW[0], hcB[0], boxes, keys, 80, 80, 8, 16.f, 32.f, 0);
  head_decode_kernel<<<dim3((NIMG * 1600 + 255) / 256), 256, 0, stream>>>(
      B(OFF_FS1), hbW[1], hbB[1], hcW[1], hcB[1], boxes, keys, 40, 40, 16, 64.f, 128.f, 12800);
  head_decode_kernel<<<dim3((NIMG * 400 + 255) / 256), 256, 0, stream>>>(
      B(OFF_FS2), hbW[2], hbB[2], hcW[2], hcB[2], boxes, keys, 20, 20, 32, 256.f, 512.f, 16000);

  // ---------------- 5) bitonic top-k sort (descending) ----------------
  for (int k = 2; k <= SORTN; k <<= 1)
    for (int j = k >> 1; j > 0; j >>= 1)
      bitonic_step_kernel<<<dim3((NIMG * SORTN) / 256), 256, 0, stream>>>(keys, j, k);

  // ---------------- 6) greedy NMS + outputs ----------------
  float* dets = (float*)d_out;
  float* keepOut = dets + (size_t)NIMG * TOPK * 5;
  nms_kernel<<<dim3(NIMG), 256, 0, stream>>>(keys, boxes, dets, keepOut);
}
